// MultiHeadedAttention_266287973008
// MI455X (gfx1250) — compile-verified, matrix operand order checked by reading, not run on hardware
//
#include <hip/hip_runtime.h>

// MI455X / gfx1250 multi-head attention, all matmuls via V_WMMA_F32_16X16X32_BF16.
// B=4, S=2048, D=1024, H=16, DK=64.
// Round 2: vectorized (b128) tile staging + GLOBAL_LOAD_ASYNC_TO_LDS for the
// pure-copy Q/K tiles in the attention kernel (ASYNCcnt path, no VGPR bounce).

#define BATCH 4
#define SEQ   2048
#define DMODEL 1024
#define NHEAD 16
#define HDIM  64

typedef unsigned short bf16_t;
typedef __bf16 v16bf __attribute__((ext_vector_type(16)));
typedef float  v8f   __attribute__((ext_vector_type(8)));

__device__ inline bf16_t f2bf(float f) {
  unsigned int u = __builtin_bit_cast(unsigned int, f);
  unsigned int r = u + 0x7FFFu + ((u >> 16) & 1u);   // round-nearest-even
  return (bf16_t)(r >> 16);
}
__device__ inline unsigned int pack2bf(float lo, float hi) {
  return (unsigned int)f2bf(lo) | ((unsigned int)f2bf(hi) << 16);
}
__device__ inline void store_out(float* p, float v)  { *p = v; }
__device__ inline void store_out(bf16_t* p, float v) { *p = f2bf(v); }

union FragU { uint4 u[2]; v16bf v; };

// Load a 16-element bf16 WMMA operand row for this lane.
// ISA layout (16-bit operand): element e -> k = (e>>3)*16 + half*8 + (e&7)
// => two contiguous 8-short (16B) chunks at kbase+half*8 and kbase+half*8+16.
__device__ inline v16bf load_frag(const bf16_t* row, int kbase, int half) {
  FragU f;
  const bf16_t* p = row + kbase + half * 8;
  f.u[0] = *(const uint4*)(p);
  f.u[1] = *(const uint4*)(p + 16);
  return f.v;
}

__device__ inline v8f wmma_bf16(v16bf a, v16bf b, v8f c) {
  return __builtin_amdgcn_wmma_f32_16x16x32_bf16(
      /*neg_a=*/false, a, /*neg_b=*/false, b,
      /*c_mod=*/(short)0, c, /*reuse_a=*/false, /*reuse_b=*/false);
}

// Async 16-byte global -> LDS copy (GV addressing: vdst(LDS off), v[addr64], off).
// Tracked by ASYNCcnt; pair with s_wait_asynccnt.
__device__ inline void async_cp16(unsigned int ldsOff, const void* gaddr) {
  asm volatile("global_load_async_to_lds_b128 %0, %1, off"
               :: "v"(ldsOff), "v"((unsigned long long)gaddr) : "memory");
}
__device__ inline void wait_async0() {
  asm volatile("s_wait_asynccnt 0" ::: "memory");
}

// ---------------------------------------------------------------------------
// GEMM: Y[M,N] = X[M,K] * W[K,N] + bias  (X fp32 or bf16, W/bias fp32,
// Y fp32 or bf16).  Block tile 128x128, K-step 32, 8 waves (4 M x 2 N),
// each wave owns a 32x64 subtile = 2x4 WMMA accumulators.
// ---------------------------------------------------------------------------

// A-tile staging (128 rows x 32 k), row-major [r][k] in LDS.
__device__ inline void stage_A(const float* __restrict__ X, bf16_t* As,
                               int t, int m0, int K, int kt) {
#pragma unroll
  for (int i = 0; i < 4; ++i) {                 // 1024 float4 chunks
    int idx4 = i * 256 + t;
    int r = idx4 >> 3, c4 = (idx4 & 7) * 4;     // 8 float4 per row
    float4 f = *(const float4*)&X[(size_t)(m0 + r) * K + kt + c4];
    uint2 p; p.x = pack2bf(f.x, f.y); p.y = pack2bf(f.z, f.w);
    *(uint2*)&As[r * 32 + c4] = p;              // ds_store_b64
  }
}
__device__ inline void stage_A(const bf16_t* __restrict__ X, bf16_t* As,
                               int t, int m0, int K, int kt) {
#pragma unroll
  for (int i = 0; i < 2; ++i) {                 // 512 chunks of 8 shorts
    int idx8 = i * 256 + t;
    int r = idx8 >> 2, c8 = (idx8 & 3) * 8;     // 4 chunks per row
    *(uint4*)&As[r * 32 + c8] =
        *(const uint4*)&X[(size_t)(m0 + r) * K + kt + c8];   // b128 copy
  }
}

template <typename TIN, typename TOUT>
__global__ __launch_bounds__(256) void gemm_bias_kernel(
    const TIN* __restrict__ X, const float* __restrict__ W,
    const float* __restrict__ bias, TOUT* __restrict__ Y,
    int M, int N, int K) {
  __shared__ bf16_t As[128 * 32];   // [row][k]     8 KB
  __shared__ bf16_t Bs[128 * 32];   // [col][k]     8 KB (K-contiguous per col)

  const int t    = threadIdx.x;
  const int wave = t >> 5, lane = t & 31;
  const int lr   = lane & 15, half = lane >> 4;
  const int waveM = wave & 3, waveN = wave >> 2;
  const int m0 = blockIdx.y * 128, n0 = blockIdx.x * 128;

  v8f acc[2][4] = {};

  for (int kt = 0; kt < K; kt += 32) {
    stage_A(X, As, t, m0, K, kt);
    // B tile (32x128) transposed into [n][k] so operand loads are b128.
#pragma unroll
    for (int i = 0; i < 4; ++i) {               // 1024 float4 chunks
      int idx4 = i * 256 + t;
      int kk = idx4 >> 5, n4 = (idx4 & 31) * 4; // coalesced along n
      float4 f = *(const float4*)&W[(size_t)(kt + kk) * N + n0 + n4];
      Bs[(n4 + 0) * 32 + kk] = f2bf(f.x);
      Bs[(n4 + 1) * 32 + kk] = f2bf(f.y);
      Bs[(n4 + 2) * 32 + kk] = f2bf(f.z);
      Bs[(n4 + 3) * 32 + kk] = f2bf(f.w);
    }
    if (kt + 32 < K)  // speculative prefetch of next A tile (global_prefetch_b8)
      __builtin_prefetch(&X[(size_t)(m0 + (t >> 1)) * K + kt + 32], 0, 1);
    __syncthreads();

    v16bf afr0 = load_frag(&As[(waveM * 32 + lr) * 32], 0, half);
    v16bf afr1 = load_frag(&As[(waveM * 32 + 16 + lr) * 32], 0, half);
#pragma unroll
    for (int nt = 0; nt < 4; ++nt) {
      v16bf bfr = load_frag(&Bs[(waveN * 64 + nt * 16 + lr) * 32], 0, half);
      acc[0][nt] = wmma_bf16(afr0, bfr, acc[0][nt]);
      acc[1][nt] = wmma_bf16(afr1, bfr, acc[1][nt]);
    }
    __syncthreads();
  }

  // Epilogue: D-matrix layout -> element v is row (v + 8*half), col = lane&15.
#pragma unroll
  for (int mt = 0; mt < 2; ++mt) {
#pragma unroll
    for (int nt = 0; nt < 4; ++nt) {
      int col = n0 + waveN * 64 + nt * 16 + lr;
      float bv = bias[col];
      int rbase = m0 + waveM * 32 + mt * 16 + half * 8;
#pragma unroll
      for (int v = 0; v < 8; ++v)
        store_out(&Y[(size_t)(rbase + v) * N + col], acc[mt][nt][v] + bv);
    }
  }
}

// ---------------------------------------------------------------------------
// Flash attention over one (b,h) pair and one 128-row Q tile per block.
// 8 waves; wave w owns Q rows [w*16, w*16+16).  K/V streamed in 128-key
// tiles through LDS.  Q and K tiles are pure copies -> async-to-LDS DMA.
// Online softmax in f32; P re-staged through per-wave LDS to become the
// A operand of the P*V WMMAs.
// ---------------------------------------------------------------------------
__global__ __launch_bounds__(256) void attn_kernel(
    const bf16_t* __restrict__ Q, const bf16_t* __restrict__ Kv,
    const bf16_t* __restrict__ V, bf16_t* __restrict__ O) {
  __shared__ bf16_t smem[32768];          // 64 KB; sole LDS object -> base 0
  bf16_t* Ks  = smem;                     // [key(128)][dk(64)]   16 KB
  bf16_t* VsT = smem + 8192;              // [dk(64)][key(128)]   16 KB
  bf16_t* Pw  = smem + 16384;             // 8 waves x [16][128]  32 KB (Q staged here first)

  const int t = threadIdx.x, wave = t >> 5, lane = t & 31;
  const int lr = lane & 15, half = lane >> 4;
  const int bh = blockIdx.y, b = bh >> 4, h = bh & 15;
  const int q0 = blockIdx.x * 128;
  const size_t headOff = (size_t)b * SEQ * DMODEL + (size_t)h * HDIM;

  // LDS byte offset of a pointer inside smem (smem is at LDS base).
  const char* smemBase = (const char*)smem;

  // ---- Stage Q tile (128x64) into the P region via async DMA ----
  bf16_t* Qs = Pw;
  {
    const bf16_t* gq = Q + headOff + (size_t)q0 * DMODEL;
#pragma unroll
    for (int i = 0; i < 4; ++i) {               // 1024 chunks of 16B
      int idx8 = i * 256 + t;
      int r = idx8 >> 3, c8 = (idx8 & 7) * 8;   // 8 chunks per 64-short row
      unsigned ldsOff = (unsigned)((const char*)&Qs[r * 64 + c8] - smemBase);
      async_cp16(ldsOff, gq + (size_t)r * DMODEL + c8);
    }
    wait_async0();
  }
  __syncthreads();
  v16bf qfr0 = load_frag(&Qs[(wave * 16 + lr) * 64], 0, half);
  v16bf qfr1 = load_frag(&Qs[(wave * 16 + lr) * 64], 32, half);
  __syncthreads();   // everyone done with Q before P overwrites this region

  float m_i[8], l_i[8];
#pragma unroll
  for (int v = 0; v < 8; ++v) { m_i[v] = -1e30f; l_i[v] = 0.0f; }
  v8f oacc[4] = {};
  bf16_t* Pmine = Pw + wave * 2048;

  for (int kt = 0; kt < SEQ; kt += 128) {
    // ---- K tile: async DMA straight into LDS [key][dk] ----
    {
      const bf16_t* gk = Kv + headOff + (size_t)kt * DMODEL;
#pragma unroll
      for (int i = 0; i < 4; ++i) {
        int idx8 = i * 256 + t;
        int r = idx8 >> 3, c8 = (idx8 & 7) * 8;
        unsigned ldsOff = (unsigned)((const char*)&Ks[r * 64 + c8] - smemBase);
        async_cp16(ldsOff, gk + (size_t)r * DMODEL + c8);
      }
    }
    // ---- V tile: wide loads, transposed scatter into [dk][key] ----
    {
      const bf16_t* gv = V + headOff + (size_t)kt * DMODEL;
#pragma unroll
      for (int i = 0; i < 4; ++i) {
        int idx8 = i * 256 + t;
        int r = idx8 >> 3, c8 = (idx8 & 7) * 8;
        union { uint4 u; bf16_t s[8]; } d;
        d.u = *(const uint4*)&gv[(size_t)r * DMODEL + c8];
#pragma unroll
        for (int j = 0; j < 8; ++j) VsT[(c8 + j) * 128 + r] = d.s[j];
      }
    }
    wait_async0();
    __syncthreads();

    // Scores: S = Q(16x64) * K^T(64x128) -> 8 N-tiles, 2 K-steps each.
    v8f s[8];
#pragma unroll
    for (int nt = 0; nt < 8; ++nt) {
      v8f z = {};
      v16bf b0 = load_frag(&Ks[(nt * 16 + lr) * 64], 0, half);
      z = wmma_bf16(qfr0, b0, z);
      v16bf b1 = load_frag(&Ks[(nt * 16 + lr) * 64], 32, half);
      s[nt] = wmma_bf16(qfr1, b1, z);
    }
#pragma unroll
    for (int nt = 0; nt < 8; ++nt)
#pragma unroll
      for (int v = 0; v < 8; ++v) s[nt][v] *= 0.125f;   // 1/sqrt(64)

    // Online softmax.  Element v of every acc = local row (v + 8*half);
    // the 16 lanes of a half hold the 16 columns of each N-tile.
#pragma unroll
    for (int v = 0; v < 8; ++v) {
      float mx = -1e30f;
#pragma unroll
      for (int nt = 0; nt < 8; ++nt) mx = fmaxf(mx, s[nt][v]);
      mx = fmaxf(mx, __shfl_xor(mx, 1, 32));
      mx = fmaxf(mx, __shfl_xor(mx, 2, 32));
      mx = fmaxf(mx, __shfl_xor(mx, 4, 32));
      mx = fmaxf(mx, __shfl_xor(mx, 8, 32));
      float mn = fmaxf(m_i[v], mx);
      float alpha = __expf(m_i[v] - mn);
      m_i[v] = mn;
      float rs = 0.0f;
#pragma unroll
      for (int nt = 0; nt < 8; ++nt) {
        float p = __expf(s[nt][v] - mn);
        s[nt][v] = p;
        rs += p;
      }
      rs += __shfl_xor(rs, 1, 32);
      rs += __shfl_xor(rs, 2, 32);
      rs += __shfl_xor(rs, 4, 32);
      rs += __shfl_xor(rs, 8, 32);
      l_i[v] = l_i[v] * alpha + rs;
#pragma unroll
      for (int ont = 0; ont < 4; ++ont) oacc[ont][v] *= alpha;
    }

    // Stage P (16x128 bf16) to per-wave LDS in row-major [row][key].
#pragma unroll
    for (int nt = 0; nt < 8; ++nt)
#pragma unroll
      for (int v = 0; v < 8; ++v)
        Pmine[(v + 8 * half) * 128 + nt * 16 + lr] = f2bf(s[nt][v]);
    asm volatile("s_wait_dscnt 0" ::: "memory");  // P visible to own-wave reads

    // O += P(16x128) * V(128x64): 4 K-steps x 4 N-tiles.
#pragma unroll
    for (int kki = 0; kki < 4; ++kki) {
      v16bf pfr = load_frag(&Pmine[lr * 128], kki * 32, half);
#pragma unroll
      for (int ont = 0; ont < 4; ++ont) {
        v16bf vfr = load_frag(&VsT[(ont * 16 + lr) * 128], kki * 32, half);
        oacc[ont] = wmma_bf16(pfr, vfr, oacc[ont]);
      }
    }
    __syncthreads();
  }

  // Normalize by l and write bf16 output at [b, row, h*64 + dk].
#pragma unroll
  for (int ont = 0; ont < 4; ++ont) {
    int dk = ont * 16 + lr;
#pragma unroll
    for (int v = 0; v < 8; ++v) {
      int row = q0 + wave * 16 + v + 8 * half;
      float val = oacc[ont][v] / l_i[v];
      O[headOff + (size_t)row * DMODEL + dk] = f2bf(val);
    }
  }
}

// ---------------------------------------------------------------------------
extern "C" void kernel_launch(void* const* d_in, const int* in_sizes, int n_in,
                              void* d_out, int out_size, void* d_ws, size_t ws_size,
                              hipStream_t stream) {
  const float* query = (const float*)d_in[0];
  const float* key   = (const float*)d_in[1];
  const float* value = (const float*)d_in[2];
  const float* Wq = (const float*)d_in[3];  const float* bq = (const float*)d_in[4];
  const float* Wk = (const float*)d_in[5];  const float* bk = (const float*)d_in[6];
  const float* Wv = (const float*)d_in[7];  const float* bv = (const float*)d_in[8];
  const float* Wo = (const float*)d_in[9];  const float* bo = (const float*)d_in[10];

  const int M = BATCH * SEQ;        // 8192
  const size_t MD = (size_t)M * DMODEL;
  bf16_t* qb = (bf16_t*)d_ws;       // 16 MB each
  bf16_t* kb = qb + MD;
  bf16_t* vb = kb + MD;
  bf16_t* xb = vb + MD;

  dim3 gGemm(DMODEL / 128, M / 128); // (8, 64)
  dim3 blk(256);

  gemm_bias_kernel<float, bf16_t><<<gGemm, blk, 0, stream>>>(query, Wq, bq, qb, M, DMODEL, DMODEL);
  gemm_bias_kernel<float, bf16_t><<<gGemm, blk, 0, stream>>>(key,   Wk, bk, kb, M, DMODEL, DMODEL);
  gemm_bias_kernel<float, bf16_t><<<gGemm, blk, 0, stream>>>(value, Wv, bv, vb, M, DMODEL, DMODEL);

  dim3 gAttn(SEQ / 128, BATCH * NHEAD); // (16, 64)
  attn_kernel<<<gAttn, blk, 0, stream>>>(qb, kb, vb, xb);

  gemm_bias_kernel<bf16_t, float><<<gGemm, blk, 0, stream>>>(xb, Wo, bo, (float*)d_out, M, DMODEL, DMODEL);
}
